// hybrid_loss_70403103916559
// MI455X (gfx1250) — compile-verified, matrix operand order checked by reading, not run on hardware
//
#include <hip/hip_runtime.h>

typedef __attribute__((ext_vector_type(8))) int   v8i;
typedef __attribute__((ext_vector_type(4))) float f4;
typedef __attribute__((ext_vector_type(4))) int   i4;

#define SMOOTH 1e-5f

// ---------------------------------------------------------------------------
// Kernel 0: zero the 12 global counters (t_sum[4], p_sum[4], inter[4])
// ---------------------------------------------------------------------------
__global__ void zero_counts_kernel(unsigned* __restrict__ c) {
  if (threadIdx.x < 12u) c[threadIdx.x] = 0u;
}

// ---------------------------------------------------------------------------
// Kernel 1: streaming argmax + WMMA-based class histogram.
// Each lane handles 8 consecutive pixels per iteration; a wave's 32 lanes x 8
// dwords fill one 16x64 IU8 A-matrix (256 pixels). B[k][n] = ((k&3)==(n&3))
// so D column c accumulates the count of one-hot bytes of class c.
// ---------------------------------------------------------------------------
__global__ __launch_bounds__(256) void hybrid_count_kernel(
    const float* __restrict__ pred,
    const int*   __restrict__ lab,
    unsigned*    __restrict__ counts,
    unsigned N, unsigned nthreads)
{
  const unsigned tid  = blockIdx.x * 256u + threadIdx.x;
  const unsigned lane = threadIdx.x & 31u;
  const unsigned wv   = threadIdx.x >> 5;

  const float* __restrict__ p0 = pred;
  const float* __restrict__ p1 = pred + (size_t)N;
  const float* __restrict__ p2 = pred + 2 * (size_t)N;
  const float* __restrict__ p3 = pred + 3 * (size_t)N;

  // Constant B matrix: dword value 1<<(8*(col&3)) in every B VGPR of a lane.
  const int bv = (int)(1u << ((lane & 3u) * 8u));
  const v8i B = { bv, bv, bv, bv, bv, bv, bv, bv };

  v8i accT = {0,0,0,0,0,0,0,0};   // true one-hot counts
  v8i accP = {0,0,0,0,0,0,0,0};   // pred one-hot counts
  v8i accI = {0,0,0,0,0,0,0,0};   // intersection counts

#pragma unroll
  for (unsigned it = 0u; it < 2u; ++it) {
    const unsigned pix = (tid + it * nthreads) * 8u;
    const bool ok = (pix + 8u) <= N;                       // full-chunk validity
    const unsigned cb = ok ? pix : ((N > 8u) ? ((N - 8u) & ~7u) : 0u);

    // 10 x b128 nontemporal streaming loads (4 logit planes + labels)
    f4 x0a = __builtin_nontemporal_load((const f4*)(p0 + cb));
    f4 x0b = __builtin_nontemporal_load((const f4*)(p0 + cb + 4));
    f4 x1a = __builtin_nontemporal_load((const f4*)(p1 + cb));
    f4 x1b = __builtin_nontemporal_load((const f4*)(p1 + cb + 4));
    f4 x2a = __builtin_nontemporal_load((const f4*)(p2 + cb));
    f4 x2b = __builtin_nontemporal_load((const f4*)(p2 + cb + 4));
    f4 x3a = __builtin_nontemporal_load((const f4*)(p3 + cb));
    f4 x3b = __builtin_nontemporal_load((const f4*)(p3 + cb + 4));
    i4 l0  = __builtin_nontemporal_load((const i4*)(lab + cb));
    i4 l1  = __builtin_nontemporal_load((const i4*)(lab + cb + 4));

    v8i At, Ap, Ai;
#pragma unroll
    for (int j = 0; j < 8; ++j) {
      const float v0 = (j < 4) ? x0a[j] : x0b[j - 4];
      const float v1 = (j < 4) ? x1a[j] : x1b[j - 4];
      const float v2 = (j < 4) ? x2a[j] : x2b[j - 4];
      const float v3 = (j < 4) ? x3a[j] : x3b[j - 4];
      const unsigned t = (unsigned)((j < 4) ? l0[j] : l1[j - 4]) & 3u;

      // argmax(first-occurrence) -> one-hot byte, no index materialized
      float m = v0; unsigned ph = 1u;
      bool c1 = v1 > m; m = c1 ? v1 : m; ph = c1 ? 0x00000100u : ph;
      bool c2 = v2 > m; m = c2 ? v2 : m; ph = c2 ? 0x00010000u : ph;
      bool c3 = v3 > m; m = c3 ? v3 : m; ph = c3 ? 0x01000000u : ph;

      unsigned th = 1u << (t << 3);
      th = ok ? th : 0u;
      ph = ok ? ph : 0u;
      At[j] = (int)th;
      Ap[j] = (int)ph;
      Ai[j] = (int)(th & ph);   // same class <=> overlapping one-hot byte
    }

    // Matrix-unit histogram accumulation: 3 wmmas per 256 pixels per wave.
    accT = __builtin_amdgcn_wmma_i32_16x16x64_iu8(false, At, false, B, accT, false, false);
    accP = __builtin_amdgcn_wmma_i32_16x16x64_iu8(false, Ap, false, B, accP, false, false);
    accI = __builtin_amdgcn_wmma_i32_16x16x64_iu8(false, Ai, false, B, accI, false, false);
  }

  // Sum the 8 D rows held by this lane; class c lives in lanes c and c+16.
  int sT = accT[0]+accT[1]+accT[2]+accT[3]+accT[4]+accT[5]+accT[6]+accT[7];
  int sP = accP[0]+accP[1]+accP[2]+accP[3]+accP[4]+accP[5]+accP[6]+accP[7];
  int sI = accI[0]+accI[1]+accI[2]+accI[3]+accI[4]+accI[5]+accI[6]+accI[7];

  __shared__ unsigned red[8][12];
#pragma unroll
  for (int c = 0; c < 4; ++c) {
    int tC = __shfl(sT, c, 32) + __shfl(sT, c + 16, 32);
    int pC = __shfl(sP, c, 32) + __shfl(sP, c + 16, 32);
    int iC = __shfl(sI, c, 32) + __shfl(sI, c + 16, 32);
    if (lane == 0u) {
      red[wv][c]     = (unsigned)tC;
      red[wv][4 + c] = (unsigned)pC;
      red[wv][8 + c] = (unsigned)iC;
    }
  }
  __syncthreads();

  if (threadIdx.x < 12u) {
    unsigned s = 0u;
#pragma unroll
    for (int w = 0; w < 8; ++w) s += red[w][threadIdx.x];
    atomicAdd(&counts[threadIdx.x], s);
  }
}

// ---------------------------------------------------------------------------
// Kernel 2: evaluate the scalar dice/iou formula from the 12 counts.
// ---------------------------------------------------------------------------
__global__ void finalize_kernel(const unsigned* __restrict__ c,
                                float* __restrict__ out) {
  if (threadIdx.x != 0u || blockIdx.x != 0u) return;
  float t[4], p[4], in[4];
  float total = 0.0f;
#pragma unroll
  for (int k = 0; k < 4; ++k) {
    t[k]  = (float)c[k];
    p[k]  = (float)c[4 + k];
    in[k] = (float)c[8 + k];
    total += t[k] + p[k];            // == p1h.sum() + t1h.sum()
  }
  float dsum = 0.0f, isum = 0.0f;
  int n = 0;
#pragma unroll
  for (int k = 0; k < 4; ++k) {
    if (t[k] > 0.0f) {
      ++n;
      dsum += 1.0f - (2.0f * in[k] + SMOOTH) / (p[k] + t[k] + SMOOTH);
      isum += 1.0f - (in[k] + SMOOTH) / (total + SMOOTH - in[k]);
    }
  }
  const float nf = (n > 0) ? (float)n : 1.0f;
  out[0] = dsum / nf + isum / nf;
}

// ---------------------------------------------------------------------------
extern "C" void kernel_launch(void* const* d_in, const int* in_sizes, int n_in,
                              void* d_out, int out_size, void* d_ws, size_t ws_size,
                              hipStream_t stream) {
  const float* pred   = (const float*)d_in[0];
  const int*   lab    = (const int*)d_in[1];
  float*       out    = (float*)d_out;
  unsigned*    counts = (unsigned*)d_ws;

  const unsigned N = (unsigned)in_sizes[1];          // H*W pixels
  const unsigned threads_needed = (N + 15u) / 16u;   // 16 px per thread (2 x 8)
  const unsigned blocks = (threads_needed + 255u) / 256u;
  const unsigned nthreads = blocks * 256u;

  zero_counts_kernel<<<1, 32, 0, stream>>>(counts);
  hybrid_count_kernel<<<blocks, 256, 0, stream>>>(pred, lab, counts, N, nthreads);
  finalize_kernel<<<1, 32, 0, stream>>>(counts, out);
}
